// GenDR_21852793602150
// MI455X (gfx1250) — matrix operand order repaired and checked by reading, not compiled, not run on hardware
//
#include <hip/hip_runtime.h>
#include <hip/hip_bf16.h>

typedef __attribute__((ext_vector_type(16))) _Float16 v16h;
typedef __attribute__((ext_vector_type(8)))  _Float16 v8h;
typedef __attribute__((ext_vector_type(8)))  float    v8f;

#define S        256
#define NF       256
#define GAMMA_I  1000.0f      // 1/GAMMA
#define EPS_RGB  0.001f
#define NEARZ    1.0f
#define FARZ     100.0f
#define INV_2DS  50.0f        // 1/(2*DIST_SCALE)

struct FaceC {
    float Aw[3], Bw[3], Cw[3];   // wk = Cw + px*Aw + py*Bw (pre-divided by area2)
    float ax[3], ay[3];          // edge-k start vertex
    float abx[3], aby[3];        // edge-k vector
    float il2[3];                // 1/(|ab|^2 + 1e-12)
    float iz[3];                 // 1/vz[k]
    float pad;                   // -> 28 floats
};

__device__ __forceinline__ float clamp01(float v) {
    return fminf(fmaxf(v, 0.0f), 1.0f);
}

// prob / zbar for one face at (px,py)
__device__ __forceinline__ void face_eval(const FaceC& fc, float px, float py,
                                          float& prob, float& zbar) {
    float wsum = 0.0f, zinv = 0.0f, d2min = 3.4e38f;
    bool inside = true;
#pragma unroll
    for (int k = 0; k < 3; ++k) {
        float w = fmaf(py, fc.Bw[k], fmaf(px, fc.Aw[k], fc.Cw[k]));
        inside = inside && (w >= 0.0f);
        float wc = clamp01(w);
        wsum += wc;
        zinv = fmaf(wc, fc.iz[k], zinv);
        float apx = px - fc.ax[k];
        float apy = py - fc.ay[k];
        float dot = fmaf(apx, fc.abx[k], apy * fc.aby[k]);
        float t = clamp01(dot * fc.il2[k]);
        float dx = fmaf(-t, fc.abx[k], apx);
        float dy = fmaf(-t, fc.aby[k], apy);
        float d2 = fmaf(dx, dx, dy * dy);
        d2min = fminf(d2min, d2);
    }
    float d = sqrtf(d2min + 1e-20f);
    float sgn = inside ? d : -d;
    prob = clamp01(fmaf(sgn, INV_2DS, 0.5f));
    float zp = wsum / (zinv + 1e-12f);
    zp = fminf(fmaxf(zp, NEARZ), FARZ);
    zbar = (FARZ - zp) * (1.0f / (FARZ - NEARZ));
}

__global__ __launch_bounds__(256)
void softras_kernel(const float* __restrict__ fv,     // (B,NF,3,3)
                    const float* __restrict__ ftex,   // (B,NF,1,3)
                    float* __restrict__ out) {        // (B,4,S,S)
    __shared__ __align__(16) FaceC    faces[NF];
    __shared__ __align__(16) _Float16 ewTile[8][32][32];   // [wave][pixel][face]
    __shared__ __align__(16) _Float16 colLDS[NF][4];
    __shared__ float                  denomT[8][32];

    const int tid  = threadIdx.x;
    const int lane = tid & 31;
    const int wv   = tid >> 5;
    const int bIdx = blockIdx.z;
    const int tx0  = blockIdx.x * 16;
    const int ty0  = blockIdx.y * 16;

    // ---- cooperative per-face precompute (one face per thread) ----
    {
        const int f = tid;
        const float* vp = fv + ((size_t)(bIdx * NF + f)) * 9;
        float vx[3], vy[3], vz[3];
#pragma unroll
        for (int v = 0; v < 3; ++v) {
            vx[v] = vp[v * 3 + 0];
            vy[v] = vp[v * 3 + 1];
            vz[v] = vp[v * 3 + 2];
        }
        float area2 = (vx[1] - vx[0]) * (vy[2] - vy[0]) -
                      (vy[1] - vy[0]) * (vx[2] - vx[0]);
        if (fabsf(area2) < 1e-12f) area2 = 1e-12f;
        float ia = 1.0f / area2;
        FaceC fc;
#pragma unroll
        for (int k = 0; k < 3; ++k) {
            int a = (k + 1) % 3, b2 = (k + 2) % 3;
            fc.Cw[k] = (vx[a] * vy[b2] - vy[a] * vx[b2]) * ia;
            fc.Aw[k] = (vy[a] - vy[b2]) * ia;
            fc.Bw[k] = (vx[b2] - vx[a]) * ia;
            int ib = (k + 1) % 3;
            fc.ax[k]  = vx[k];
            fc.ay[k]  = vy[k];
            fc.abx[k] = vx[ib] - vx[k];
            fc.aby[k] = vy[ib] - vy[k];
            fc.il2[k] = 1.0f / (fc.abx[k] * fc.abx[k] + fc.aby[k] * fc.aby[k] + 1e-12f);
            fc.iz[k]  = 1.0f / vz[k];
        }
        fc.pad = 0.0f;
        faces[f] = fc;
        const float* tp = ftex + ((size_t)(bIdx * NF + f)) * 3;
        colLDS[f][0] = (_Float16)tp[0];
        colLDS[f][1] = (_Float16)tp[1];
        colLDS[f][2] = (_Float16)tp[2];
        colLDS[f][3] = (_Float16)0.0f;
    }
    __syncthreads();

    // ---- pixel for this lane: wave covers rows ty0+2w, ty0+2w+1 ----
    const int py_i = ty0 + 2 * wv + (lane >> 4);
    const int px_i = tx0 + (lane & 15);
    const float px = (2.0f * (float)px_i + 1.0f) * (1.0f / (float)S) - 1.0f;
    const float py = -((2.0f * (float)py_i + 1.0f) * (1.0f / (float)S) - 1.0f);

    // ---- pass 1: zmax and alpha product ----
    float zmax = 0.0f;
    float aprod = 1.0f;
#pragma unroll 1
    for (int f = 0; f < NF; ++f) {
        float prob, zbar;
        face_eval(faces[f], px, py, prob, zbar);
        zmax = fmaxf(zmax, (prob > 0.0f) ? zbar : 0.0f);
        aprod *= (1.0f - prob);
    }
    zmax = fmaxf(zmax, EPS_RGB);

    // ---- pass 2: ew tiles -> WMMA accumulation ----
    v8f c0 = {};  // pixels 0..15 of this wave
    v8f c1 = {};  // pixels 16..31
    float s = 0.0f;
    const bool hi   = lane >= 16;
    const int  ncol = lane & 15;
    const int  r    = lane & 15;
    const int  koff = hi ? 8 : 0;
    const int  kb   = hi ? 16 : 0;

#pragma unroll 1
    for (int chunk = 0; chunk < NF / 32; ++chunk) {
#pragma unroll 1
        for (int j = 0; j < 32; ++j) {
            int f = chunk * 32 + j;
            float prob, zbar;
            face_eval(faces[f], px, py, prob, zbar);
            float ew = prob * __expf((zbar - zmax) * GAMMA_I);
            s += ew;
            ewTile[wv][lane][j] = (_Float16)ew;
        }
        asm volatile("s_wait_dscnt 0" ::: "memory");

        // A fragments per ISA layout (lanes 0-15: K0-7 & K16-23; lanes 16-31: K8-15 & K24-31)
        v8h lo0 = *(const v8h*)&ewTile[wv][r][koff];
        v8h hi0 = *(const v8h*)&ewTile[wv][r][koff + 16];
        v8h lo1 = *(const v8h*)&ewTile[wv][r + 16][koff];
        v8h hi1 = *(const v8h*)&ewTile[wv][r + 16][koff + 16];
        v16h a0, a1;
#pragma unroll
        for (int i = 0; i < 8; ++i) {
            a0[i] = lo0[i]; a0[i + 8] = hi0[i];
            a1[i] = lo1[i]; a1[i + 8] = hi1[i];
        }
        // B fragment: 32x16, columns 0..2 = face colors, rest zero
        v16h bf;
#pragma unroll
        for (int h = 0; h < 16; ++h) {
            _Float16 cv = (ncol < 3) ? colLDS[chunk * 32 + koff + h + (h >= 8 ? 8 : 0)][ncol] : (_Float16)0.0f;
            bf[h] = cv;
        }
        asm volatile("s_wait_dscnt 0" ::: "memory");

        c0 = __builtin_amdgcn_wmma_f32_16x16x32_f16(false, a0, false, bf,
                                                    (short)0, c0, false, false);
        c1 = __builtin_amdgcn_wmma_f32_16x16x32_f16(false, a1, false, bf,
                                                    (short)0, c1, false, false);
    }

    const float ebg = __expf((EPS_RGB - zmax) * GAMMA_I);
    const float denom = s + ebg;
    denomT[wv][lane] = denom;

    // alpha channel (c=3), one per lane/pixel
    out[(((size_t)bIdx * 4 + 3) * S + py_i) * S + px_i] = 1.0f - aprod;

    asm volatile("s_wait_dscnt 0" ::: "memory");

    // ---- scatter WMMA result: lane holds channel `ncol`, rows M = j + (hi?8:0) ----
    if (ncol < 3) {
        const int y0 = ty0 + 2 * wv;
        const int y1 = y0 + 1;
#pragma unroll
        for (int j = 0; j < 8; ++j) {
            int m = j + (hi ? 8 : 0);            // pixel column within tile row
            float v0 = c0[j] / denomT[wv][m];
            float v1 = c1[j] / denomT[wv][16 + m];
            out[(((size_t)bIdx * 4 + ncol) * S + y0) * S + tx0 + m] = v0;
            out[(((size_t)bIdx * 4 + ncol) * S + y1) * S + tx0 + m] = v1;
        }
    }
}

extern "C" void kernel_launch(void* const* d_in, const int* in_sizes, int n_in,
                              void* d_out, int out_size, void* d_ws, size_t ws_size,
                              hipStream_t stream) {
    (void)in_sizes; (void)n_in; (void)d_ws; (void)ws_size; (void)out_size;
    const float* fv   = (const float*)d_in[0];   // face_vertices (2,256,3,3) f32
    const float* ftex = (const float*)d_in[1];   // face_textures (2,256,1,3) f32
    float* out = (float*)d_out;                  // (2,4,256,256) f32
    dim3 grid(S / 16, S / 16, 2);
    dim3 block(256);
    softras_kernel<<<grid, block, 0, stream>>>(fv, ftex, out);
}
